// acLSTM_41747082117226
// MI455X (gfx1250) — matrix operand order repaired and checked by reading, not compiled, not run on hardware
//
#include <hip/hip_runtime.h>

// Problem constants (match reference)
#define HID  1024
#define H4   4096
#define FR   171
#define FRP  192    // FRAME padded to multiple of 32 for K of cell 1
#define FRD  176    // decoder N padded to multiple of 16
#define BB   128
#define TT   256

typedef __attribute__((ext_vector_type(16))) __bf16 v16bf;
typedef __attribute__((ext_vector_type(8)))  __bf16 v8bf;
typedef __attribute__((ext_vector_type(8)))  float  v8f;

// ---------- helpers ----------

__device__ __forceinline__ __bf16 f2bf(float f) {
    union { float f; unsigned u; } in; in.f = f;
    unsigned u = in.u;
    unsigned r = (u + 0x7FFFu + ((u >> 16) & 1u)) >> 16; // RNE
    union { unsigned short s; __bf16 b; } out; out.s = (unsigned short)r;
    return out.b;
}

__device__ __forceinline__ float sigmoidf_(float x) { return 1.0f / (1.0f + __expf(-x)); }

// A-matrix fragment (16x32 bf16, row-major source X[ld]):
// lane<16 : row=lane,    elements e -> K=k0+e
// lane>=16: row=lane-16, elements e -> K=k0+16+e
__device__ __forceinline__ v16bf load_a_frag(const __bf16* X, int ld, int m0, int k0, int lane) {
    const int r  = lane & 15;
    const int kh = (lane >> 4) << 4;            // 0 or 16
    const __bf16* p = X + (size_t)(m0 + r) * ld + (k0 + kh);
    union { v16bf v; v8bf h[2]; } u;
    u.h[0] = *(const v8bf*)(p);
    u.h[1] = *(const v8bf*)(p + 8);
    return u.v;
}

// B-matrix fragment (32x16 bf16). For z = x @ W^T, B[k][n] = W[n][k], W row-major [Nrows][ld].
// lane<16 : n=lane,    elements e -> K=k0+e
// lane>=16: n=lane-16, elements e -> K=k0+16+e   (32-byte contiguous per lane)
__device__ __forceinline__ v16bf load_b_frag(const __bf16* W, int ld, int n0, int k0, int lane) {
    const int n  = lane & 15;
    const int kh = (lane >> 4) << 4;            // 0 or 16
    return *(const v16bf*)(W + (size_t)(n0 + n) * ld + (k0 + kh));
}

// ---------- preprocessing kernels ----------

__global__ void convert_pad_kernel(const float* __restrict__ src, __bf16* __restrict__ dst,
                                   int srows, int scols, int drows, int dcols) {
    int idx = blockIdx.x * blockDim.x + threadIdx.x;
    if (idx >= drows * dcols) return;
    int r = idx / dcols, c = idx % dcols;
    float v = (r < srows && c < scols) ? src[(size_t)r * scols + c] : 0.0f;
    dst[idx] = f2bf(v);
}

__global__ void bias_sum_kernel(const float* __restrict__ a, const float* __restrict__ b,
                                float* __restrict__ o, int n) {
    int i = blockIdx.x * blockDim.x + threadIdx.x;
    if (i < n) o[i] = a[i] + b[i];
}

// Build step-t input: auto-conditioning mux (ground truth vs previous model output), bf16, padded.
__global__ void prep_input_kernel(const float* __restrict__ seq, const float* __restrict__ outbuf,
                                  __bf16* __restrict__ xin,
                                  const int* __restrict__ gtp, const int* __restrict__ condp, int t) {
    int idx = blockIdx.x * blockDim.x + threadIdx.x;
    if (idx >= BB * FRP) return;
    int b = idx / FRP, c = idx % FRP;
    float v = 0.0f;
    if (c < FR) {
        int gt = gtp[0], cond = condp[0];
        int per = gt + cond; if (per <= 0) per = 1;
        bool use_gt = (t % per) < gt;
        if (use_gt)      v = seq[((size_t)b * TT + t) * FR + c];
        else if (t > 0)  v = outbuf[((size_t)b * TT + (t - 1)) * FR + c];
    }
    xin[idx] = f2bf(v);
}

// ---------- fused LSTM cell ----------
// z = Xin@Wx^T + Hprev@Wh^T + bias; then gate math; c,h update.
// grid: (64 j-tiles, 4 m-slices); block: ONE wave32.
// Each wave owns 2 m-tiles x 4 gates x 1 j-tile = 8 WMMA output tiles.
// A-fragments are reused across all 4 gates; gates for a given (m,j) element live in
// the same lane, so the LSTM nonlinearity runs entirely in registers (no LDS, no barrier).
__global__ __launch_bounds__(32) void lstm_cell_kernel(
    const __bf16* __restrict__ Xin, int ldx, int KX,
    const __bf16* __restrict__ Hprev,
    const __bf16* __restrict__ Wx,     // [4096][KX] bf16
    const __bf16* __restrict__ Wh,     // [4096][1024] bf16
    const float*  __restrict__ bias,   // [4096] = b_ih + b_hh
    float*        __restrict__ Cst,    // [128][1024] f32 (in/out)
    __bf16*       __restrict__ Hout)   // [128][1024] bf16 (distinct from Hprev)
{
    const int lane = threadIdx.x;
    const int j0   = blockIdx.x * 16;         // hidden-unit tile
    const int m0   = blockIdx.y * 32;         // 2 batch-row tiles

    v8f acc[2][4];
    #pragma unroll
    for (int mt = 0; mt < 2; ++mt)
        #pragma unroll
        for (int g = 0; g < 4; ++g)
            acc[mt][g] = (v8f){0.f,0.f,0.f,0.f,0.f,0.f,0.f,0.f};

    for (int k0 = 0; k0 < KX; k0 += 32) {
        v16bf a0 = load_a_frag(Xin, ldx, m0,      k0, lane);
        v16bf a1 = load_a_frag(Xin, ldx, m0 + 16, k0, lane);
        #pragma unroll
        for (int g = 0; g < 4; ++g) {
            v16bf b = load_b_frag(Wx, KX, g * HID + j0, k0, lane);
            acc[0][g] = __builtin_amdgcn_wmma_f32_16x16x32_bf16(false, a0, false, b, (short)0, acc[0][g], false, false);
            acc[1][g] = __builtin_amdgcn_wmma_f32_16x16x32_bf16(false, a1, false, b, (short)0, acc[1][g], false, false);
        }
    }
    for (int k0 = 0; k0 < HID; k0 += 32) {
        v16bf a0 = load_a_frag(Hprev, HID, m0,      k0, lane);
        v16bf a1 = load_a_frag(Hprev, HID, m0 + 16, k0, lane);
        #pragma unroll
        for (int g = 0; g < 4; ++g) {
            v16bf b = load_b_frag(Wh, HID, g * HID + j0, k0, lane);
            acc[0][g] = __builtin_amdgcn_wmma_f32_16x16x32_bf16(false, a0, false, b, (short)0, acc[0][g], false, false);
            acc[1][g] = __builtin_amdgcn_wmma_f32_16x16x32_bf16(false, a1, false, b, (short)0, acc[1][g], false, false);
        }
    }

    // C/D layout: VGPR r -> row r (lanes 0-15) or row r+8 (lanes 16-31); col = lane&15.
    // All 4 gate tiles share this layout, so the cell update is pure register math.
    const int col = lane & 15;
    const int rb  = (lane >> 4) << 3;
    const int jg  = j0 + col;
    const float bi = bias[0 * HID + jg];
    const float bf = bias[1 * HID + jg];
    const float bg = bias[2 * HID + jg];
    const float bo = bias[3 * HID + jg];

    #pragma unroll
    for (int mt = 0; mt < 2; ++mt) {
        #pragma unroll
        for (int r = 0; r < 8; ++r) {
            int m = m0 + mt * 16 + rb + r;
            size_t p = (size_t)m * HID + jg;
            float zi = acc[mt][0][r] + bi;
            float zf = acc[mt][1][r] + bf;
            float zg = acc[mt][2][r] + bg;
            float zo = acc[mt][3][r] + bo;
            float cold = Cst[p];
            float cnew = sigmoidf_(zf) * cold + sigmoidf_(zi) * tanhf(zg);
            Cst[p]  = cnew;
            Hout[p] = f2bf(sigmoidf_(zo) * tanhf(cnew));
        }
    }
}

// ---------- decoder: out[b,t,:] = h3 @ dec_w^T + dec_b ----------
__global__ __launch_bounds__(256) void decoder_kernel(
    const __bf16* __restrict__ H3,     // [128][1024] bf16
    const __bf16* __restrict__ Wd,     // [176][1024] bf16 (rows >=171 zero)
    const float*  __restrict__ bd,     // [171]
    float* __restrict__ out, int t)
{
    const int lane = threadIdx.x;
    const int mt   = threadIdx.y;      // 0..7 -> batch rows mt*16..
    const int n0   = blockIdx.x * 16;  // 0..10 tiles
    v8f acc = {0.f, 0.f, 0.f, 0.f, 0.f, 0.f, 0.f, 0.f};
    for (int k0 = 0; k0 < HID; k0 += 32) {
        v16bf a = load_a_frag(H3, HID, mt * 16, k0, lane);
        v16bf b = load_b_frag(Wd, HID, n0, k0, lane);
        acc = __builtin_amdgcn_wmma_f32_16x16x32_bf16(false, a, false, b, (short)0, acc, false, false);
    }
    const int col = lane & 15;
    const int rb  = (lane >> 4) << 3;
    const int c   = n0 + col;
    if (c < FR) {
        float bv = bd[c];
        #pragma unroll
        for (int r = 0; r < 8; ++r) {
            int m = mt * 16 + rb + r;
            out[((size_t)m * TT + t) * FR + c] = acc[r] + bv;
        }
    }
}

// ---------- host ----------

extern "C" void kernel_launch(void* const* d_in, const int* in_sizes, int n_in,
                              void* d_out, int out_size, void* d_ws, size_t ws_size,
                              hipStream_t stream) {
    (void)in_sizes; (void)n_in; (void)out_size; (void)ws_size;

    const float* seq   = (const float*)d_in[0];
    const float* w_ih1 = (const float*)d_in[1];
    const float* w_hh1 = (const float*)d_in[2];
    const float* b_ih1 = (const float*)d_in[3];
    const float* b_hh1 = (const float*)d_in[4];
    const float* w_ih2 = (const float*)d_in[5];
    const float* w_hh2 = (const float*)d_in[6];
    const float* b_ih2 = (const float*)d_in[7];
    const float* b_hh2 = (const float*)d_in[8];
    const float* w_ih3 = (const float*)d_in[9];
    const float* w_hh3 = (const float*)d_in[10];
    const float* b_ih3 = (const float*)d_in[11];
    const float* b_hh3 = (const float*)d_in[12];
    const float* dec_w = (const float*)d_in[13];
    const float* dec_b = (const float*)d_in[14];
    const int*   gtp   = (const int*)d_in[15];
    const int*   condp = (const int*)d_in[16];
    float* out = (float*)d_out;

    // workspace carve-up (~46 MB), 256B aligned
    char* ws = (char*)d_ws;
    size_t off = 0;
    auto alloc = [&](size_t bytes) -> void* {
        off = (off + 255) & ~(size_t)255;
        void* p = ws + off; off += bytes; return p;
    };
    __bf16* w1x = (__bf16*)alloc((size_t)H4 * FRP * 2);
    __bf16* w1h = (__bf16*)alloc((size_t)H4 * HID * 2);
    __bf16* w2x = (__bf16*)alloc((size_t)H4 * HID * 2);
    __bf16* w2h = (__bf16*)alloc((size_t)H4 * HID * 2);
    __bf16* w3x = (__bf16*)alloc((size_t)H4 * HID * 2);
    __bf16* w3h = (__bf16*)alloc((size_t)H4 * HID * 2);
    __bf16* wd  = (__bf16*)alloc((size_t)FRD * HID * 2);
    float*  b1  = (float*)alloc((size_t)H4 * 4);
    float*  b2  = (float*)alloc((size_t)H4 * 4);
    float*  b3  = (float*)alloc((size_t)H4 * 4);
    __bf16* h1a = (__bf16*)alloc((size_t)BB * HID * 2);
    __bf16* h1b = (__bf16*)alloc((size_t)BB * HID * 2);
    __bf16* h2a = (__bf16*)alloc((size_t)BB * HID * 2);
    __bf16* h2b = (__bf16*)alloc((size_t)BB * HID * 2);
    __bf16* h3a = (__bf16*)alloc((size_t)BB * HID * 2);
    __bf16* h3b = (__bf16*)alloc((size_t)BB * HID * 2);
    float*  c1  = (float*)alloc((size_t)BB * HID * 4);
    float*  c2  = (float*)alloc((size_t)BB * HID * 4);
    float*  c3  = (float*)alloc((size_t)BB * HID * 4);
    __bf16* xin = (__bf16*)alloc((size_t)BB * FRP * 2);

    const int thr = 256;
    // bf16 weight conversion (idempotent, captured in graph)
    convert_pad_kernel<<<(H4 * FRP + thr - 1) / thr, thr, 0, stream>>>(w_ih1, w1x, H4, FR, H4, FRP);
    convert_pad_kernel<<<(H4 * HID + thr - 1) / thr, thr, 0, stream>>>(w_hh1, w1h, H4, HID, H4, HID);
    convert_pad_kernel<<<(H4 * HID + thr - 1) / thr, thr, 0, stream>>>(w_ih2, w2x, H4, HID, H4, HID);
    convert_pad_kernel<<<(H4 * HID + thr - 1) / thr, thr, 0, stream>>>(w_hh2, w2h, H4, HID, H4, HID);
    convert_pad_kernel<<<(H4 * HID + thr - 1) / thr, thr, 0, stream>>>(w_ih3, w3x, H4, HID, H4, HID);
    convert_pad_kernel<<<(H4 * HID + thr - 1) / thr, thr, 0, stream>>>(w_hh3, w3h, H4, HID, H4, HID);
    convert_pad_kernel<<<(FRD * HID + thr - 1) / thr, thr, 0, stream>>>(dec_w, wd, FR, HID, FRD, HID);
    bias_sum_kernel<<<(H4 + thr - 1) / thr, thr, 0, stream>>>(b_ih1, b_hh1, b1, H4);
    bias_sum_kernel<<<(H4 + thr - 1) / thr, thr, 0, stream>>>(b_ih2, b_hh2, b2, H4);
    bias_sum_kernel<<<(H4 + thr - 1) / thr, thr, 0, stream>>>(b_ih3, b_hh3, b3, H4);

    // zero initial states every call (workspace is not re-poisoned between replays)
    hipMemsetAsync(h1a, 0, (size_t)BB * HID * 2, stream);
    hipMemsetAsync(h1b, 0, (size_t)BB * HID * 2, stream);
    hipMemsetAsync(h2a, 0, (size_t)BB * HID * 2, stream);
    hipMemsetAsync(h2b, 0, (size_t)BB * HID * 2, stream);
    hipMemsetAsync(h3a, 0, (size_t)BB * HID * 2, stream);
    hipMemsetAsync(h3b, 0, (size_t)BB * HID * 2, stream);
    hipMemsetAsync(c1, 0, (size_t)BB * HID * 4, stream);
    hipMemsetAsync(c2, 0, (size_t)BB * HID * 4, stream);
    hipMemsetAsync(c3, 0, (size_t)BB * HID * 4, stream);

    const dim3 cellGrid(HID / 16, 4);   // 64 j-tiles x 4 m-slices (32 rows each)
    const dim3 cellBlk(32);             // one wave32 per block
    const dim3 decGrid(FRD / 16);       // 11 n-tiles
    const dim3 decBlk(32, 8);

    for (int t = 0; t < TT; ++t) {
        __bf16* h1p = (t & 1) ? h1b : h1a;  __bf16* h1n = (t & 1) ? h1a : h1b;
        __bf16* h2p = (t & 1) ? h2b : h2a;  __bf16* h2n = (t & 1) ? h2a : h2b;
        __bf16* h3p = (t & 1) ? h3b : h3a;  __bf16* h3n = (t & 1) ? h3a : h3b;

        prep_input_kernel<<<(BB * FRP + thr - 1) / thr, thr, 0, stream>>>(seq, out, xin, gtp, condp, t);
        lstm_cell_kernel<<<cellGrid, cellBlk, 0, stream>>>(xin, FRP, FRP, h1p, w1x, w1h, b1, c1, h1n);
        lstm_cell_kernel<<<cellGrid, cellBlk, 0, stream>>>(h1n, HID, HID, h2p, w2x, w2h, b2, c2, h2n);
        lstm_cell_kernel<<<cellGrid, cellBlk, 0, stream>>>(h2n, HID, HID, h3p, w3x, w3h, b3, c3, h3n);
        decoder_kernel<<<decGrid, decBlk, 0, stream>>>(h3n, wd, dec_b, out, t);
    }
}
